// CausalAttentionHead_17463337026257
// MI455X (gfx1250) — matrix operand order repaired and checked by reading, not compiled, not run on hardware
//
#include <hip/hip_runtime.h>

// Causal attention head, B=4, T=4096, D=1024, H=64 (fp32 in/out).
// f32->f16 weight transpose, WMMA f16 QKV projection with TDM-staged
// (tensor_load_to_lds, double-buffered) weight tiles, WMMA flash attention
// with online softmax, balanced causal scheduling, next-chunk prefetch.

typedef _Float16 h16;
typedef __attribute__((ext_vector_type(16))) _Float16 v16h;
typedef __attribute__((ext_vector_type(8)))  _Float16 v8h;
typedef __attribute__((ext_vector_type(8)))  float    v8f;
typedef __attribute__((ext_vector_type(4)))  float    v4f;

typedef __attribute__((ext_vector_type(4))) unsigned int tdm_v4u;
typedef __attribute__((ext_vector_type(8))) int          tdm_v8i;
typedef __attribute__((ext_vector_type(4))) int          tdm_v4i;

#if defined(__has_builtin)
#if __has_builtin(__builtin_amdgcn_tensor_load_to_lds)
#define USE_TDM 1
#endif
#endif
#ifndef USE_TDM
#define USE_TDM 0
#endif

#define WMMA_F16(A, B, C) \
  __builtin_amdgcn_wmma_f32_16x16x32_f16(false, (A), false, (B), (short)0, (C), false, false)

static __device__ __forceinline__ v16h joinv(v8h lo, v8h hi) {
  v16h r;
#pragma unroll
  for (int i = 0; i < 8; ++i) { r[i] = lo[i]; r[i + 8] = hi[i]; }
  return r;
}
static __device__ __forceinline__ float redmax16(float v) {
#pragma unroll
  for (int m = 1; m < 16; m <<= 1) v = fmaxf(v, __shfl_xor(v, m, 32));
  return v;
}
static __device__ __forceinline__ float redsum16(float v) {
#pragma unroll
  for (int m = 1; m < 16; m <<= 1) v += __shfl_xor(v, m, 32);
  return v;
}

#if USE_TDM
// LDS aperture: flat addr[31:0] == LDS byte address.
static __device__ __forceinline__ unsigned int lds_addr32(const void* p) {
  return (unsigned int)(uintptr_t)p;
}
// TDM 2D tile load: tile 32(k) x 192(rows) of f16 from Wt [192][1024] into LDS
// (rows packed contiguously, 64B per row). D# layout per CDNA5 ISA ch.8.
// Toolchain builtin arity = 6: (g0 v4u, g1 v8i, g2 v4i, g3 v4i, v8i, cpol).
static __device__ __forceinline__ void tdm_load_w(unsigned int lds_byte,
                                                  const h16* Wt, int kc) {
  unsigned long long ga = (unsigned long long)(uintptr_t)(Wt + (kc << 5));
  tdm_v4u g0;
  g0[0] = 1u;                                    // count=1 valid, user, load
  g0[1] = lds_byte;                              // D#.lds_addr (bytes)
  g0[2] = (unsigned int)ga;                      // global_addr[31:0]
  g0[3] = (unsigned int)((ga >> 32) & 0x01FFFFFFu) | (2u << 30); // [56:32] | type=2
  tdm_v8i g1;
  g1[0] = 1 << 16;                               // data_size=1 -> 2-byte elems
  g1[1] = (int)(1024u << 16);                    // tensor_dim0[15:0]
  g1[2] = (int)(192u << 16);                     // tensor_dim0 hi | tensor_dim1 lo
  g1[3] = (int)(32u << 16);                      // tensor_dim1 hi | tile_dim0=32
  g1[4] = 192;                                   // tile_dim1=192, tile_dim2=0
  g1[5] = 1024;                                  // tensor_dim0_stride (elems)
  g1[6] = 0;
  g1[7] = 0;
  tdm_v4i gz4 = {0, 0, 0, 0};
  tdm_v8i gz8 = {0, 0, 0, 0, 0, 0, 0, 0};
  __builtin_amdgcn_tensor_load_to_lds(g0, g1, gz4, gz4, gz8, 0);
}
#endif

// ---- Kernel 0: weights f32 [1024][64] -> f16 transposed [64][1024]; [q,k,v].
__global__ __launch_bounds__(256) void wcvt(const float* __restrict__ Wq,
                                            const float* __restrict__ Wk,
                                            const float* __restrict__ Wv,
                                            h16* __restrict__ Wt) {
  int idx = blockIdx.x * 256 + threadIdx.x;
  if (idx >= 3 * 65536) return;
  int t = idx >> 16, rem = idx & 65535;
  int n = rem >> 10, k = rem & 1023;
  const float* W = (t == 0) ? Wq : ((t == 1) ? Wk : Wv);
  Wt[idx] = (h16)W[k * 64 + n];
}

// ---- Kernel 1: QKV projection. One wave: 16 rows x 64 cols of Q,K,V.
// Weight chunks staged in LDS via TDM, double-buffered.
__global__ __launch_bounds__(128) void qkv_project(const float* __restrict__ x,
                                                   const h16* __restrict__ Wt,
                                                   h16* __restrict__ Qh,
                                                   h16* __restrict__ Kh,
                                                   h16* __restrict__ Vt) {
#if USE_TDM
  __shared__ __align__(128) h16 wbuf[2][192 * 32];   // 2 x 12KB
#endif
  const int lane = threadIdx.x & 31;
  const int wave = threadIdx.x >> 5;
  const int l16  = lane & 15;
  const bool hiH = lane >= 16;
  const int hi8  = hiH ? 8 : 0;
  const int rowBlk = blockIdx.x * 4 + wave;          // 0..1023
  const int row0 = rowBlk << 4;

  v8f acc[3][4] = {};
  const float* xrow = x + (size_t)(row0 + l16) * 1024;

#if USE_TDM
  if (wave == 0) tdm_load_w(lds_addr32(&wbuf[0][0]), Wt, 0);
#endif

  for (int kc = 0; kc < 32; ++kc) {
    const int kb = kc << 5;
#if USE_TDM
    if (wave == 0) {
      if (kc + 1 < 32) {
        tdm_load_w(lds_addr32(&wbuf[(kc + 1) & 1][0]), Wt, kc + 1);
        __builtin_amdgcn_s_wait_tensorcnt((short)1);   // chunk kc done (in-order)
      } else {
        __builtin_amdgcn_s_wait_tensorcnt((short)0);
      }
    }
    __syncthreads();                                   // data-ready broadcast
#endif
    // A fragment (16x32 f16): lanes 0-15 -> K kb+0..15,
    // lanes 16-31 -> K kb+16..31; row M = lane%16.
    const int ka = kb + (hiH ? 16 : 0);
    v4f a0 = *(const v4f*)(xrow + ka);
    v4f a1 = *(const v4f*)(xrow + ka + 4);
    v4f a2 = *(const v4f*)(xrow + ka + 8);
    v4f a3 = *(const v4f*)(xrow + ka + 12);
    v16h af;
#pragma unroll
    for (int i = 0; i < 4; ++i) {
      af[i] = (h16)a0[i]; af[i + 4] = (h16)a1[i];
      af[i + 8] = (h16)a2[i]; af[i + 12] = (h16)a3[i];
    }
    // B fragments (32x16): lane n<16 holds K kb..kb+15 of col n, lanes 16-31
    // hold K kb+16..kb+31 of col n-16; contiguous 16 f16 per lane.
    // Load ALL fragments first (distinct regs -> pipelined ds_loads),
    // then issue the 12 WMMAs back-to-back.
    v16h bf[12];
#if USE_TDM
    const h16* wchunk = &wbuf[kc & 1][0];
#pragma unroll
    for (int t = 0; t < 3; ++t)
#pragma unroll
      for (int nt = 0; nt < 4; ++nt)
        bf[t * 4 + nt] =
            *(const v16h*)(wchunk + (t * 64 + nt * 16 + l16) * 32 + (hiH ? 16 : 0));
#else
    const int wk = kb + (hiH ? 16 : 0);
#pragma unroll
    for (int t = 0; t < 3; ++t)
#pragma unroll
      for (int nt = 0; nt < 4; ++nt)
        bf[t * 4 + nt] =
            *(const v16h*)(Wt + (((size_t)(t * 64 + nt * 16 + l16) << 10) + wk));
#endif
#pragma unroll
    for (int t = 0; t < 3; ++t)
#pragma unroll
      for (int nt = 0; nt < 4; ++nt)
        acc[t][nt] = WMMA_F16(af, bf[t * 4 + nt], acc[t][nt]);
#if USE_TDM
    __syncthreads();                                   // reads done before overwrite
#endif
  }

  // Store. C/D layout: VGPR r -> row r (lanes 0-15) / r+8 (lanes 16-31), col=lane%16.
  const int bidx = row0 >> 12;
  const int t0 = row0 & 4095;
#pragma unroll
  for (int nt = 0; nt < 4; ++nt) {
    const int col = nt * 16 + l16;
#pragma unroll
    for (int r = 0; r < 8; ++r) {
      const int m = r + hi8;
      const size_t o = (size_t)(row0 + m) * 64 + col;
      Qh[o] = (h16)acc[0][nt][r];
      Kh[o] = (h16)acc[1][nt][r];
    }
    v8h pk;
#pragma unroll
    for (int r = 0; r < 8; ++r) pk[r] = (h16)acc[2][nt][r];
    *(v8h*)(Vt + (((size_t)(bidx * 64 + col)) << 12) + t0 + hi8) = pk;
  }
}

// ---- Kernel 2: flash attention. One wave per 16-query block, 32-key chunks.
// q-blocks remapped in complementary pairs (j, 255-j) per block of 4 waves so
// causal work per block is uniform (~129 chunks per pair).
__global__ __launch_bounds__(128) void attn(const h16* __restrict__ Qh,
                                            const h16* __restrict__ Kh,
                                            const h16* __restrict__ Vt,
                                            float* __restrict__ out) {
  __shared__ __align__(32) h16 pst[4][16][32];   // per-wave P restage tile
  const int lane = threadIdx.x & 31;
  const int wave = threadIdx.x >> 5;
  const int l16  = lane & 15;
  const bool hiH = lane >= 16;
  const int hi8  = hiH ? 8 : 0;
  const int g = blockIdx.x * 4 + wave;            // 0..1023
  const int b = g >> 8;
  const int i = g & 255;
  const int qb = (i & 1) ? (255 - (i >> 1)) : (i >> 1);  // balanced causal map
  const int qrow0 = (b << 12) + (qb << 4);

  const h16* qrow = Qh + (size_t)(qrow0 + l16) * 64;
  v16h qf[2];
#pragma unroll
  for (int f = 0; f < 2; ++f) {
    const int off = f * 32 + (hiH ? 16 : 0);
    qf[f] = joinv(*(const v8h*)(qrow + off), *(const v8h*)(qrow + off + 8));
  }

  v8f o[4] = {};
  float mrow[8], lrow[8];
#pragma unroll
  for (int r = 0; r < 8; ++r) { mrow[r] = -__builtin_inff(); lrow[r] = 0.f; }

  const float scale = 0.125f;                     // H^-0.5
  const int nch = (qb >> 1) + 1;                  // causal
  const h16* kbat = Kh + ((size_t)(b) << 12) * 64;
  const h16* vbat = Vt + ((size_t)(b * 64) << 12);

  for (int c = 0; c < nch; ++c) {
    const int k0 = c << 5;
    // Prefetch next chunk's K rows and V rows (global_prefetch_b8).
    if (c + 1 < nch) {
      const int kn = (c + 1) << 5;
      __builtin_prefetch(kbat + (size_t)(kn + lane) * 64, 0, 1);
      __builtin_prefetch(vbat + ((size_t)lane << 12) + kn, 0, 1);
      __builtin_prefetch(vbat + ((size_t)(lane + 32) << 12) + kn, 0, 1);
    }
    // Batch-load K fragments (2 key tiles x 2 headdim chunks) and V fragments
    // (4 head tiles) up front: V latency hides behind the softmax chain.
    v16h kf[4];
#pragma unroll
    for (int j = 0; j < 2; ++j) {
      const h16* krow = kbat + (size_t)(k0 + j * 16 + l16) * 64 + (hiH ? 16 : 0);
      kf[j * 2 + 0] = *(const v16h*)krow;
      kf[j * 2 + 1] = *(const v16h*)(krow + 32);
    }
    v16h vf[4];
#pragma unroll
    for (int nt = 0; nt < 4; ++nt)
      vf[nt] = *(const v16h*)(vbat + (((size_t)(nt * 16 + l16)) << 12) + k0 +
                              (hiH ? 16 : 0));
    // S = Q K^T for two 16-key tiles.
    v8f s[2];
#pragma unroll
    for (int j = 0; j < 2; ++j) {
      v8f t = {};
      t = WMMA_F16(qf[0], kf[j * 2 + 0], t);
      t = WMMA_F16(qf[1], kf[j * 2 + 1], t);
      s[j] = t;
    }
    // Causal mask, scale, online softmax.
    float p0[8], p1[8], alpha[8];
#pragma unroll
    for (int r = 0; r < 8; ++r) {
      const int m = r + hi8;
      const int qt = (qb << 4) + m;
      const float s0 = ((k0 + l16) > qt)      ? -__builtin_inff() : s[0][r] * scale;
      const float s1 = ((k0 + 16 + l16) > qt) ? -__builtin_inff() : s[1][r] * scale;
      const float mx = redmax16(fmaxf(s0, s1));
      const float mn = fmaxf(mrow[r], mx);
      alpha[r] = __expf(mrow[r] - mn);
      mrow[r] = mn;
      p0[r] = __expf(s0 - mn);
      p1[r] = __expf(s1 - mn);
      lrow[r] = lrow[r] * alpha[r] + redsum16(p0[r] + p1[r]);
    }
#pragma unroll
    for (int nt = 0; nt < 4; ++nt)
#pragma unroll
      for (int r = 0; r < 8; ++r) o[nt][r] *= alpha[r];

    // Restage P (C-layout f32) -> LDS f16 -> A-layout fragment.
#pragma unroll
    for (int r = 0; r < 8; ++r) {
      const int m = r + hi8;
      pst[wave][m][l16]      = (h16)p0[r];
      pst[wave][m][16 + l16] = (h16)p1[r];
    }
    __threadfence_block();   // wave-private tile; per-wave LDS in-order
    const int off = hiH ? 16 : 0;
    v16h pa = joinv(*(const v8h*)&pst[wave][l16][off],
                    *(const v8h*)&pst[wave][l16][off + 8]);

    // O += P V  (V transposed: lane n = head dim, contiguous keys).
#pragma unroll
    for (int nt = 0; nt < 4; ++nt)
      o[nt] = WMMA_F16(pa, vf[nt], o[nt]);
  }

  float inv[8];
#pragma unroll
  for (int r = 0; r < 8; ++r) inv[r] = 1.0f / lrow[r];
#pragma unroll
  for (int nt = 0; nt < 4; ++nt) {
    const int col = nt * 16 + l16;
#pragma unroll
    for (int r = 0; r < 8; ++r)
      out[(size_t)(qrow0 + r + hi8) * 64 + col] = o[nt][r] * inv[r];
  }
}

extern "C" void kernel_launch(void* const* d_in, const int* in_sizes, int n_in,
                              void* d_out, int out_size, void* d_ws, size_t ws_size,
                              hipStream_t stream) {
  // setup_inputs dict order: x, Wk, Wq, Wv
  const float* x  = (const float*)d_in[0];
  const float* Wk = (const float*)d_in[1];
  const float* Wq = (const float*)d_in[2];
  const float* Wv = (const float*)d_in[3];
  float* out = (float*)d_out;

  // Workspace (f16): Qh 2MB | Kh 2MB | Vt 2MB | Wt 384KB
  h16* Qh = (h16*)d_ws;
  h16* Kh = Qh + (size_t)16384 * 64;
  h16* Vt = Kh + (size_t)16384 * 64;
  h16* Wt = Vt + (size_t)4 * 64 * 4096;

  wcvt<<<(3 * 65536) / 256, 256, 0, stream>>>(Wq, Wk, Wv, Wt);
  qkv_project<<<256, 128, 0, stream>>>(x, Wt, Qh, Kh, Vt);
  attn<<<256, 128, 0, stream>>>(Qh, Kh, Vt, out);
}
